// ParallelWaveganGenerator_55327768708127
// MI455X (gfx1250) — matrix operand order repaired and checked by reading, not compiled, not run on hardware
//
#include <hip/hip_runtime.h>

// ============================================================================
// ParallelWaveGAN generator for MI455X (gfx1250, wave32, WMMA).
//
// Roofline: ~1.3 TFLOP total, but ~19 GB of activation traffic if fully
// HBM-resident => memory bound. We therefore (a) keep activations in bf16
// (halves bytes, enables v_wmma_f32_16x16x32_bf16), (b) use a channel-
// contiguous [b][t][C] layout so WMMA B-fragments are contiguous 32B/lane
// loads, (c) run layer-by-layer so h (65MB bf16) + cu (98MB bf16) stay
// mostly resident in the 192MB L2, (d) stage activation tiles into LDS with
// the CDNA5 async global->LDS path (ASYNCcnt) so staging never touches the
// register file.
//
// WMMA mapping (D = A*B + C, 16x16x32 bf16):
//   A = weight tile (16 Cout x 32 Cin), packed into a contiguous-K
//       A-fragment layout (lane<16 holds K=0..15 of row
//       M=lane; lane>=16 holds K=16..31 of row M=lane-16).
//   B = activation tile (32 Cin x 16 timesteps): lane = kh*16+n holds
//       K = kh*16+0..15 of column n -> 16 contiguous bf16 per lane.
//   C/D = f32 16x16 (Cout x time) accumulator.
// ============================================================================

typedef __attribute__((ext_vector_type(16))) __bf16 v16bf;
typedef __attribute__((ext_vector_type(8)))  float  v8f;

#define T_LEN  256000
#define BATCH  2
#define NBLK   30

__device__ __forceinline__ v8f wmma_bf16(v16bf a, v16bf b, v8f c) {
    return __builtin_amdgcn_wmma_f32_16x16x32_bf16(
        /*neg_a=*/false, a, /*neg_b=*/false, b,
        /*c_mod=*/(short)0, c, /*reuse_a=*/false, /*reuse_b=*/false);
}

// Async 16B global->LDS copy (GLOBAL_LOAD_ASYNC_TO_LDS_B128, ASYNCcnt).
// ldsaddr: wave-relative LDS byte address = low 32 bits of the generic
// pointer to a __shared__ object (ISA 10.2: LDS aperture keeps the LDS
// offset in addr[31:0]).
__device__ __forceinline__ void async_copy_b128(const void* gptr, void* sptr) {
    unsigned ldsaddr = (unsigned)(uintptr_t)sptr;
    asm volatile("global_load_async_to_lds_b128 %0, %1, off"
                 :: "v"(ldsaddr), "v"(gptr) : "memory");
}
__device__ __forceinline__ void wait_asynccnt0() {
    asm volatile("s_wait_asynccnt 0" ::: "memory");
}

// ---------------------------------------------------------------------------
// Weight packing: f32 weights -> bf16 WMMA A-fragments.
// Fragment element j (0..15) of lane L maps to
//   K_local = j + (L>>4)*16,  M = (L&15).
// ---------------------------------------------------------------------------

// conv_w (30,128,64,3) -> apc[l][tap][mt(8)][kf(2)][lane(32)][16]
__global__ void pack_conv_kernel(const float* __restrict__ w, __bf16* __restrict__ dst) {
    int idx = blockIdx.x * 256 + threadIdx.x;
    if (idx >= 30 * 3 * 8 * 2 * 512) return;
    int j    = idx & 15;
    int lane = (idx >> 4) & 31;
    int kf   = (idx >> 9) & 1;
    int mt   = (idx >> 10) & 7;
    int tap  = (idx >> 13) % 3;
    int l    = idx / 24576;
    int M = mt * 16 + (lane & 15);
    int K = kf * 32 + j + ((lane >> 4) << 4);
    dst[idx] = (__bf16)w[(((size_t)l * 128 + M) * 64 + K) * 3 + tap];
}

// aux_w (30,128,80) padded K->96 -> apa[l][mt(8)][kf(3)][lane][16]
__global__ void pack_aux_kernel(const float* __restrict__ w, __bf16* __restrict__ dst) {
    int idx = blockIdx.x * 256 + threadIdx.x;
    if (idx >= 30 * 8 * 3 * 512) return;
    int j    = idx & 15;
    int lane = (idx >> 4) & 31;
    int kf   = (idx >> 9) % 3;
    int mt   = (idx / 1536) & 7;
    int l    = idx / 12288;
    int M = mt * 16 + (lane & 15);
    int K = kf * 32 + j + ((lane >> 4) << 4);
    float v = (K < 80) ? w[((size_t)l * 128 + M) * 80 + K] : 0.0f;
    dst[idx] = (__bf16)v;
}

// skip_w / res_w (30,64,64) -> ap[l][mt(4)][kf(2)][lane][16]
__global__ void pack64_kernel(const float* __restrict__ w, __bf16* __restrict__ dst) {
    int idx = blockIdx.x * 256 + threadIdx.x;
    if (idx >= 30 * 4 * 2 * 512) return;
    int j    = idx & 15;
    int lane = (idx >> 4) & 31;
    int kf   = (idx >> 9) & 1;
    int mt   = (idx >> 10) & 3;
    int l    = idx >> 12;
    int M = mt * 16 + (lane & 15);
    int K = kf * 32 + j + ((lane >> 4) << 4);
    dst[idx] = (__bf16)w[((size_t)l * 64 + M) * 64 + K];
}

// ---------------------------------------------------------------------------
// Upsample net: nearest-neighbor x4 then k=9 conv (pad 4), 4 stages.
// out[n] = sum_k w[k] * rep[n+k-4],  rep[j] = in[j/4], zero-padded.
// Stages 0..2 keep [b][c][t] f32; final stage emits cu as [b][t][96] bf16
// (channels 80..95 zero-padded so the aux GEMM K-dim is a clean 3x32).
// ---------------------------------------------------------------------------
__global__ void up_stage_kernel(const float* __restrict__ in, float* __restrict__ out,
                                const float* __restrict__ w, int Tin, int total) {
    int idx = blockIdx.x * 256 + threadIdx.x;
    if (idx >= total) return;
    int Tout = Tin * 4;
    int t  = idx % Tout;
    int bc = idx / Tout;
    float acc = 0.0f;
#pragma unroll
    for (int k = 0; k < 9; ++k) {
        int j = t + k - 4;
        if (j >= 0 && j < Tout) acc += w[k] * in[(size_t)bc * Tin + (j >> 2)];
    }
    out[(size_t)bc * Tout + t] = acc;
}

__global__ void up_final_kernel(const float* __restrict__ in, __bf16* __restrict__ cu,
                                const float* __restrict__ w, int total) {
    int idx = blockIdx.x * 256 + threadIdx.x;
    if (idx >= total) return;
    int c = idx % 96;
    size_t bt = (size_t)idx / 96;          // b*T + t
    int t = (int)(bt % T_LEN);
    int b = (int)(bt / T_LEN);
    float acc = 0.0f;
    if (c < 80) {
#pragma unroll
        for (int k = 0; k < 9; ++k) {
            int j = t + k - 4;
            if (j >= 0 && j < T_LEN) acc += w[k] * in[((size_t)b * 80 + c) * 64000 + (j >> 2)];
        }
    }
    cu[idx] = (__bf16)acc;
}

// ---------------------------------------------------------------------------
// Head: h[b][t][c] = first_w[c] * x[b][t] + first_b[c]   (bf16 out)
// ---------------------------------------------------------------------------
__global__ void first_kernel(const float* __restrict__ x, const float* __restrict__ w,
                             const float* __restrict__ bias, __bf16* __restrict__ h, int total) {
    int idx = blockIdx.x * 256 + threadIdx.x;
    if (idx >= total) return;
    int c = idx & 63;
    size_t bt = (size_t)idx >> 6;
    h[idx] = (__bf16)(w[c] * x[bt] + bias[c]);
}

__global__ void zero_f32_kernel(float* __restrict__ p, int total) {
    int idx = blockIdx.x * 256 + threadIdx.x;
    if (idx < total) p[idx] = 0.0f;
}

// ---------------------------------------------------------------------------
// One WaveNet residual block over a 16-timestep tile (per workgroup).
//   stage    : h taps (3x16x64) + cu (16x96) async-copied global->LDS
//   waves 0..7 : 16-row slice of the 128-row gated conv
//                (3 dilated taps x 2 K-frags) + aux (3 K-frags)  = 9 WMMAs
//   gate via LDS: z = tanh(g[0:64]) * sigmoid(g[64:128])
//   waves 0..3 : skip GEMM (+= into f32 skip accumulator)        = 2 WMMAs
//   waves 4..7 : res GEMM  (h_out = (res(z)+h_in)*sqrt(0.5))     = 2 WMMAs
// ---------------------------------------------------------------------------
__global__ __launch_bounds__(256)
void wavenet_block_kernel(const __bf16* __restrict__ h_in, __bf16* __restrict__ h_out,
                          const __bf16* __restrict__ cu, float* __restrict__ skip,
                          const __bf16* __restrict__ apc, const __bf16* __restrict__ apa,
                          const __bf16* __restrict__ aps, const __bf16* __restrict__ apr,
                          const float* __restrict__ conv_b, const float* __restrict__ skip_b,
                          const float* __restrict__ res_b, int l, int d) {
    const int tid  = threadIdx.x;
    const int wave = tid >> 5;
    const int lane = tid & 31;
    const int tile = blockIdx.x;                 // 0 .. B*T/16-1
    const int b    = tile / (T_LEN / 16);
    const int t0   = (tile % (T_LEN / 16)) * 16;

    __shared__ __align__(32) __bf16 hts[3][16][64];   // 3 dilated taps, bf16
    __shared__ __align__(32) __bf16 cuts[16][96];     // aux conditioning tile
    __shared__ float  gs[16][128];                    // pre-gate activations
    __shared__ __align__(32) __bf16 zs[16][64];       // gated output

    // ---- async-stage activation tiles into LDS (16B chunks) ----
    // In-range chunks: GLOBAL_LOAD_ASYNC_TO_LDS_B128 (no VGPR round trip).
    // Out-of-range halo chunks: ds_store of zeros (disjoint LDS addresses,
    // so no ordering hazard between the DS and ASYNC pipes).
    for (int idx = tid; idx < 3 * 16 * 8; idx += 256) {
        int chunk = idx & 7;                 // 8 bf16 = 16B
        int tl    = (idx >> 3) & 15;
        int tap   = idx >> 7;
        int t     = t0 + tl + (tap - 1) * d;
        if (t >= 0 && t < T_LEN) {
            async_copy_b128(h_in + ((size_t)(b * T_LEN + t) * 64 + chunk * 8),
                            &hts[tap][tl][chunk * 8]);
        } else {
            uint4 z = {0u, 0u, 0u, 0u};
            *(uint4*)(&hts[tap][tl][chunk * 8]) = z;
        }
    }
    for (int idx = tid; idx < 16 * 12; idx += 256) {
        int chunk = idx % 12, tl = idx / 12;
        async_copy_b128(cu + ((size_t)(b * T_LEN + t0 + tl) * 96 + chunk * 8),
                        &cuts[tl][chunk * 8]);
    }
    wait_asynccnt0();
    __syncthreads();

    const int n  = lane & 15;                // timestep column within tile
    const int lh = lane >> 4;                // K half-wave

    // ---- gated conv: rows [16*wave, 16*wave+16) of the 128 outputs ----
    v8f acc = {};
    {
        const __bf16* apcl = apc + (size_t)l * (3 * 8 * 2) * 512;
#pragma unroll
        for (int tap = 0; tap < 3; ++tap)
#pragma unroll
            for (int kf = 0; kf < 2; ++kf) {
                v16bf a = *(const v16bf*)(apcl + (size_t)(((tap * 8 + wave) * 2 + kf)) * 512 + lane * 16);
                v16bf bb = *(const v16bf*)(&hts[tap][n][kf * 32 + lh * 16]);
                acc = wmma_bf16(a, bb, acc);
            }
        const __bf16* apal = apa + (size_t)l * (8 * 3) * 512;
#pragma unroll
        for (int kf = 0; kf < 3; ++kf) {
            v16bf a = *(const v16bf*)(apal + (size_t)(wave * 3 + kf) * 512 + lane * 16);
            v16bf bb = *(const v16bf*)(&cuts[n][kf * 32 + lh * 16]);
            acc = wmma_bf16(a, bb, acc);
        }
    }
    // D layout: VGPR v holds M = wave*16 + lh*8 + v for column n.
#pragma unroll
    for (int v = 0; v < 8; ++v) {
        int m = wave * 16 + lh * 8 + v;
        gs[n][m] = acc[v] + conv_b[l * 128 + m];
    }
    __syncthreads();

    // ---- gate: z = tanh(ga+ca) * sigmoid(gb+cb) (aux already accumulated) --
    for (int idx = tid; idx < 16 * 64; idx += 256) {
        int m = idx & 63, nn = idx >> 6;
        float ga = gs[nn][m];
        float gb = gs[nn][m + 64];
        float z = tanhf(ga) * (1.0f / (1.0f + __expf(-gb)));
        zs[nn][m] = (__bf16)z;
    }
    __syncthreads();

    // ---- skip (waves 0..3) and residual (waves 4..7) GEMMs ----
    {
        const int  mt     = wave & 3;
        const bool is_res = wave >= 4;
        const __bf16* ap  = (is_res ? apr : aps) + ((size_t)l * 8 + mt * 2) * 512;
        v8f acc2 = {};
#pragma unroll
        for (int kf = 0; kf < 2; ++kf) {
            v16bf a  = *(const v16bf*)(ap + (size_t)kf * 512 + lane * 16);
            v16bf bb = *(const v16bf*)(&zs[n][kf * 32 + lh * 16]);
            acc2 = wmma_bf16(a, bb, acc2);
        }
        const int t     = t0 + n;
        const int mbase = mt * 16 + lh * 8;          // 8 contiguous channels
        const size_t base = (size_t)(b * T_LEN + t) * 64 + mbase;
        if (!is_res) {
            float* sp = skip + base;
#pragma unroll
            for (int v = 0; v < 8; ++v)
                sp[v] += acc2[v] + skip_b[l * 64 + mbase + v];
        } else {
            const __bf16* hp = h_in + base;
            __bf16* op = h_out + base;
#pragma unroll
            for (int v = 0; v < 8; ++v)
                op[v] = (__bf16)((acc2[v] + res_b[l * 64 + mbase + v] + (float)hp[v]) *
                                 0.70710678118654752f);
        }
    }
}

// ---------------------------------------------------------------------------
// Tail: out = w2 . relu(W1 . relu(skips*sqrt(1/30)) + b1) + b2
// ---------------------------------------------------------------------------
__global__ __launch_bounds__(256)
void last_kernel(const float* __restrict__ skip, const float* __restrict__ w1,
                 const float* __restrict__ b1, const float* __restrict__ w2,
                 const float* __restrict__ b2, float* __restrict__ out) {
    __shared__ float sw1[64 * 64];
    __shared__ float sb1[64];
    __shared__ float sw2[64];
    for (int i = threadIdx.x; i < 4096; i += 256) sw1[i] = w1[i];
    if (threadIdx.x < 64) {
        sb1[threadIdx.x] = b1[threadIdx.x];
        sw2[threadIdx.x] = w2[threadIdx.x];
    }
    __syncthreads();

    const size_t i = (size_t)blockIdx.x * 256 + threadIdx.x;   // over B*T
    const float scale = 0.18257418583505537f;                  // sqrt(1/30)
    float s[64];
    const float* sp = skip + i * 64;
#pragma unroll
    for (int c = 0; c < 64; ++c) {
        float v = sp[c] * scale;                               // relu o scale
        s[c] = v > 0.0f ? v : 0.0f;
    }
    float o = b2[0];
    for (int m = 0; m < 64; ++m) {
        float a = sb1[m];
#pragma unroll
        for (int c = 0; c < 64; ++c) a += sw1[m * 64 + c] * s[c];
        o += sw2[m] * (a > 0.0f ? a : 0.0f);
    }
    out[i] = o;
}

// ---------------------------------------------------------------------------
extern "C" void kernel_launch(void* const* d_in, const int* in_sizes, int n_in,
                              void* d_out, int out_size, void* d_ws, size_t ws_size,
                              hipStream_t stream) {
    (void)in_sizes; (void)n_in; (void)out_size; (void)ws_size;
    const float* c       = (const float*)d_in[0];
    const float* x       = (const float*)d_in[1];
    const float* first_w = (const float*)d_in[2];
    const float* first_b = (const float*)d_in[3];
    const float* up_ws   = (const float*)d_in[4];
    const float* conv_w  = (const float*)d_in[5];
    const float* conv_b  = (const float*)d_in[6];
    const float* aux_w   = (const float*)d_in[7];
    const float* skip_w  = (const float*)d_in[8];
    const float* skip_b  = (const float*)d_in[9];
    const float* res_w   = (const float*)d_in[10];
    const float* res_b   = (const float*)d_in[11];
    const float* last_w1 = (const float*)d_in[12];
    const float* last_b1 = (const float*)d_in[13];
    const float* last_w2 = (const float*)d_in[14];
    const float* last_b2 = (const float*)d_in[15];
    float* out = (float*)d_out;

    // Workspace carving (~417 MB total: cu 98MB, h x2 131MB, skip 131MB,
    // upsample temps 54MB, packed weights ~2.7MB).
    char* ws = (char*)d_ws;
    size_t off = 0;
    auto carve = [&](size_t bytes) -> char* {
        off = (off + 255) & ~(size_t)255;
        char* p = ws + off;
        off += bytes;
        return p;
    };
    __bf16* cu   = (__bf16*)carve((size_t)BATCH * T_LEN * 96 * 2);
    __bf16* hA   = (__bf16*)carve((size_t)BATCH * T_LEN * 64 * 2);
    __bf16* hB   = (__bf16*)carve((size_t)BATCH * T_LEN * 64 * 2);
    float*  skip = (float*) carve((size_t)BATCH * T_LEN * 64 * 4);
    float*  up1  = (float*) carve((size_t)BATCH * 80 * 4000 * 4);
    float*  up2  = (float*) carve((size_t)BATCH * 80 * 16000 * 4);
    float*  up3  = (float*) carve((size_t)BATCH * 80 * 64000 * 4);
    __bf16* apc  = (__bf16*)carve((size_t)NBLK * 3 * 8 * 2 * 512 * 2);
    __bf16* apa  = (__bf16*)carve((size_t)NBLK * 8 * 3 * 512 * 2);
    __bf16* aps  = (__bf16*)carve((size_t)NBLK * 4 * 2 * 512 * 2);
    __bf16* apr  = (__bf16*)carve((size_t)NBLK * 4 * 2 * 512 * 2);

    // Pack weights into WMMA A-fragment layout (cheap, done per call).
    pack_conv_kernel<<<(737280 + 255) / 256, 256, 0, stream>>>(conv_w, apc);
    pack_aux_kernel <<<(368640 + 255) / 256, 256, 0, stream>>>(aux_w, apa);
    pack64_kernel   <<<(122880 + 255) / 256, 256, 0, stream>>>(skip_w, aps);
    pack64_kernel   <<<(122880 + 255) / 256, 256, 0, stream>>>(res_w, apr);

    // Upsample net (80ch, x4 four times: 1000 -> 256000).
    up_stage_kernel<<<2500,   256, 0, stream>>>(c,   up1, up_ws + 0,  1000,  640000);
    up_stage_kernel<<<10000,  256, 0, stream>>>(up1, up2, up_ws + 9,  4000,  2560000);
    up_stage_kernel<<<40000,  256, 0, stream>>>(up2, up3, up_ws + 18, 16000, 10240000);
    up_final_kernel<<<192000, 256, 0, stream>>>(up3, cu,  up_ws + 27, 49152000);

    zero_f32_kernel<<<128000, 256, 0, stream>>>(skip, BATCH * T_LEN * 64);
    first_kernel   <<<128000, 256, 0, stream>>>(x, first_w, first_b, hA, BATCH * T_LEN * 64);

    // 30 residual blocks, ping-ponging h (global sync between dilations).
    __bf16* hin = hA;
    __bf16* hout = hB;
    for (int l = 0; l < NBLK; ++l) {
        int d = 1 << (l % 10);
        wavenet_block_kernel<<<BATCH * (T_LEN / 16), 256, 0, stream>>>(
            hin, hout, cu, skip, apc, apa, aps, apr, conv_b, skip_b, res_b, l, d);
        __bf16* tmp = hin; hin = hout; hout = tmp;
    }

    last_kernel<<<BATCH * T_LEN / 256, 256, 0, stream>>>(skip, last_w1, last_b1,
                                                         last_w2, last_b2, out);
}
